// FastLayer_30975304139510
// MI455X (gfx1250) — matrix operand order repaired and checked by reading, not compiled, not run on hardware
//
#include <hip/hip_runtime.h>

// ---------------------------------------------------------------------------
// Problem constants (match the JAX reference)
// ---------------------------------------------------------------------------
static constexpr int B_ = 4;
static constexpr int L_ = 8192;
static constexpr int H_ = 1024;
static constexpr int P_ = 64;
static constexpr int M_ = B_ * L_;          // 32768 tokens
static constexpr int K_INJ = 3 * H_;        // 3072
static constexpr int K_OUT = 4 * P_;        // 256  (xf_re | xf_im | xb_re | xb_im)
static constexpr int N_BU  = 2 * P_;        // 128  (Bu re | Bu im)

typedef __attribute__((ext_vector_type(16))) __bf16    v16bf;
typedef __attribute__((ext_vector_type(8)))  float     v8f;
typedef __attribute__((ext_vector_type(4)))  unsigned  u32x4;
typedef __attribute__((ext_vector_type(8)))  unsigned  u32x8;

// ---------------------------------------------------------------------------
// bf16 <-> f32 helpers
// ---------------------------------------------------------------------------
__device__ __forceinline__ __bf16 f2bf(float f) {
  unsigned u = __builtin_bit_cast(unsigned, f);
  unsigned r = (u + 0x7FFFu + ((u >> 16) & 1u)) >> 16;   // round-to-nearest-even
  unsigned short s = (unsigned short)r;
  return __builtin_bit_cast(__bf16, s);
}
__device__ __forceinline__ float bf2f(__bf16 h) {
  unsigned short s = __builtin_bit_cast(unsigned short, h);
  unsigned u = ((unsigned)s) << 16;
  return __builtin_bit_cast(float, u);
}

// ---------------------------------------------------------------------------
// WMMA fragment loaders (wave32 layouts per CDNA5 ISA 7.12.2)
// ---------------------------------------------------------------------------
__device__ __forceinline__ v16bf load_a_frag(const __bf16* __restrict__ A, int lda,
                                             int m0, int kb, int lane) {
  const int row  = m0 + (lane & 15);
  const int half = lane >> 4;
  const __bf16* p = A + (size_t)row * lda + kb + half * 8;
  v16bf f;
  uint4* fp = reinterpret_cast<uint4*>(&f);
  fp[0] = *reinterpret_cast<const uint4*>(p);        // elems 0..7  (K = half*8 + 0..7)
  fp[1] = *reinterpret_cast<const uint4*>(p + 16);   // elems 8..15 (K = 16 + half*8 + ..)
  return f;
}

// B tile staged in LDS, N-major, row pitch BPITCH elems (80B: 64B data + 16B TDM pad)
static constexpr int BPITCH = 40;
__device__ __forceinline__ v16bf lds_b_frag(const __bf16* sm, int n0loc, int lane) {
  const int col  = n0loc + (lane & 15);
  const int half = lane >> 4;
  const __bf16* p = sm + col * BPITCH + half * 8;    // elem i<8 -> K = half*8 + i; i>=8 -> K = 16 + half*8 + (i-8)
  v16bf f;
  uint4* fp = reinterpret_cast<uint4*>(&f);
  fp[0] = *reinterpret_cast<const uint4*>(p);
  fp[1] = *reinterpret_cast<const uint4*>(p + 16);
  return f;
}

// ---------------------------------------------------------------------------
// TDM: DMA one nrows(N) x 32(K) bf16 weight tile (row pitch Kdim*2 B in memory)
// into LDS at `ldsdst`, padding 4 DWORDs after every 16 DWORDs (80B LDS pitch).
// Descriptor per CDNA5 ISA §8.3/§8.4; 2D tensor -> 2 SGPR groups.
// ---------------------------------------------------------------------------
__device__ __forceinline__ void tdm_load_btile(const __bf16* Bt, int Kdim, int Ndim,
                                               int n0, int kb, int nrows,
                                               __bf16* ldsdst) {
  unsigned long long gaddr = (unsigned long long)(const void*)(Bt + (size_t)n0 * Kdim + kb);
  unsigned lds_off = (unsigned)(unsigned long long)(void*)ldsdst;
  u32x4 g0;
  g0[0] = 1u;                                            // count=1, user descriptor
  g0[1] = lds_off;                                       // lds_addr
  g0[2] = (unsigned)gaddr;                               // global_addr[31:0]
  g0[3] = (unsigned)((gaddr >> 32) & 0x01FFFFFFull)      // global_addr[56:32]
          | (2u << 30);                                  // type=2 ("image")
  const unsigned td0 = (unsigned)(Kdim >> 1);            // row length, 4B units
  const unsigned td1 = (unsigned)Ndim;                   // #rows in tensor
  u32x8 g1;
  g1[0] = (2u << 16)                                     // data_size = 4B
        | (1u << 20)                                     // pad_enable
        | (3u << 22)                                     // pad_interval = 16 DWORDs
        | (3u << 25);                                    // pad_amount   = 4 DWORDs
  g1[1] = (td0 & 0xFFFFu) << 16;                         // tensor_dim0[15:0]
  g1[2] = (td0 >> 16) | ((td1 & 0xFFFFu) << 16);         // tensor_dim0[31:16] | dim1[15:0]
  g1[3] = (td1 >> 16) | (16u << 16);                     // dim1[31:16] | tile_dim0 = 16 units
  g1[4] = (unsigned)nrows;                               // tile_dim1 = nrows, tile_dim2 = 0
  g1[5] = td0;                                           // tensor_dim0_stride[31:0] (units)
  g1[6] = 0u;                                            // stride[47:32] | dim1_stride[15:0]
  g1[7] = 0u;
  asm volatile("tensor_load_to_lds %0, %1" :: "s"(g0), "s"(g1) : "memory");
}

// ---------------------------------------------------------------------------
// bf16 WMMA GEMM:  C[m,n] = sum_k A[m,k] * Bt[n,k]  (+ epilogues)
// Block = 256 threads = 8 waves laid out 4(M) x 2(N).
// Block tile 128(M) x (32*WNT)(N); each wave: 32 x (16*WNT) = 2 x WNT tiles.
// B tile double-buffered in LDS via TDM; A fragments register double-buffered;
// B fragments pipelined one tile ahead.
// ---------------------------------------------------------------------------
template<int WNT, bool OUT_BF16, bool EPI_BIAS, bool EPI_DU>
__global__ __launch_bounds__(256) void wmma_gemm_kernel(
    const __bf16* __restrict__ A, const __bf16* __restrict__ Bt,
    void* __restrict__ Cout, int Mdim, int Ndim, int Kdim,
    const float* __restrict__ bias, const float* __restrict__ Dvec,
    const __bf16* __restrict__ U) {
  constexpr int BN = 32 * WNT;                       // block col extent
  __shared__ __align__(16) __bf16 bsm[2][BN * BPITCH];
  const int lane = threadIdx.x & 31;
  const int wid  = threadIdx.x >> 5;                 // 0..7
  const int wm   = wid & 3;                          // wave row 0..3
  const int wn   = wid >> 2;                         // wave col 0..1
  const int m0   = blockIdx.x * 128 + wm * 32;       // two 16-row tiles
  const int nblk = blockIdx.y * BN;
  const int nw   = wn * (16 * WNT);                  // col offset inside LDS tile
  const bool tdm_wave = (wid == 0);

  // prologue: DMA first B tile, preload first A fragments
  if (tdm_wave) tdm_load_btile(Bt, Kdim, Ndim, nblk, 0, BN, &bsm[0][0]);
  v16bf a0 = load_a_frag(A, Kdim, m0, 0, lane);
  v16bf a1 = load_a_frag(A, Kdim, m0 + 16, 0, lane);
  if (tdm_wave) __builtin_amdgcn_s_wait_tensorcnt(0);
  __syncthreads();

  v8f acc[2][WNT] = {};
  int cur = 0;
  for (int kb = 0; kb < Kdim; kb += 32) {
    const int nxt = kb + 32;
    const bool more = (nxt < Kdim);
    if (more && tdm_wave) tdm_load_btile(Bt, Kdim, Ndim, nblk, nxt, BN, &bsm[cur ^ 1][0]);
    v16bf a0n = a0, a1n = a1;
    if (more) {
      a0n = load_a_frag(A, Kdim, m0, nxt, lane);
      a1n = load_a_frag(A, Kdim, m0 + 16, nxt, lane);
      __builtin_prefetch(A + (size_t)(m0 + (lane & 15)) * Kdim + nxt + 32, 0, 1);
    }
    // pipelined B fragments: load tile t+1 while computing tile t
    v16bf b0 = lds_b_frag(&bsm[cur][0], nw, lane);
#pragma unroll
    for (int t = 0; t < WNT; ++t) {
      v16bf b1 = b0;
      if (t + 1 < WNT) b1 = lds_b_frag(&bsm[cur][0], nw + 16 * (t + 1), lane);
      acc[0][t] = __builtin_amdgcn_wmma_f32_16x16x32_bf16(
          false, a0, false, b0, (short)0, acc[0][t], false, false);
      acc[1][t] = __builtin_amdgcn_wmma_f32_16x16x32_bf16(
          false, a1, false, b0, (short)0, acc[1][t], false, false);
      b0 = b1;
    }
    if (more) {
      if (tdm_wave) __builtin_amdgcn_s_wait_tensorcnt(0);
      __syncthreads();
      a0 = a0n; a1 = a1n;
      cur ^= 1;
    }
  }

  // C/D layout: lane = {half,col}; VGPR g -> row = g + half*8
  const int colL = lane & 15;
#pragma unroll
  for (int i = 0; i < 2; ++i) {
    const int rb = m0 + i * 16 + (lane >> 4) * 8;
#pragma unroll
    for (int t = 0; t < WNT; ++t) {
      const int n = nblk + nw + 16 * t + colL;
#pragma unroll
      for (int g = 0; g < 8; ++g) {
        const int m = rb + g;
        float v = acc[i][t][g];
        if (EPI_BIAS) v += bias[n];
        if (EPI_DU)   v += Dvec[n] * bf2f(U[(size_t)m * Ndim + n]);
        if (OUT_BF16) ((__bf16*)Cout)[(size_t)m * Ndim + n] = f2bf(v);
        else          ((float*)Cout)[(size_t)m * Ndim + n]  = v;
      }
    }
  }
}

// ---------------------------------------------------------------------------
// Weight preparation kernels (fp32 -> bf16, transposed N-major layouts)
// ---------------------------------------------------------------------------
__global__ void prep_winj_kernel(const float* __restrict__ Winj, __bf16* __restrict__ Wt) {
  int o = blockIdx.x * blockDim.x + threadIdx.x;     // over 1024*3072
  int n = o / K_INJ, k = o - n * K_INJ;
  Wt[o] = f2bf(Winj[(size_t)k * H_ + n]);
}

__global__ void prep_abar_kernel(const float* __restrict__ Lre, const float* __restrict__ Lim,
                                 const float* __restrict__ logdt, float* __restrict__ Abar) {
  int p = threadIdx.x;
  if (p < P_) {
    float dt = __expf(logdt[p]);
    float er = __expf(Lre[p] * dt);
    float ph = Lim[p] * dt;
    Abar[2 * p]     = er * __cosf(ph);
    Abar[2 * p + 1] = er * __sinf(ph);
  }
}

__global__ void prep_bw_kernel(const float* __restrict__ Bre, const float* __restrict__ Bim,
                               const float* __restrict__ Abar,
                               const float* __restrict__ Lre, const float* __restrict__ Lim,
                               __bf16* __restrict__ Bwt) {
  int o = blockIdx.x * blockDim.x + threadIdx.x;     // over 128*1024
  int n = o >> 10, k = o & 1023, p = n & (P_ - 1);
  float lbr = Abar[2 * p], lbi = Abar[2 * p + 1];
  float lr = Lre[p], li = Lim[p];
  float den = lr * lr + li * li;
  float nr = lbr - 1.0f, ni = lbi;
  float cr = (nr * lr + ni * li) / den;              // (Lam_bar-1)/Lam
  float ci = (ni * lr - nr * li) / den;
  float br = Bre[(size_t)p * H_ + k], bi = Bim[(size_t)p * H_ + k];
  float re = cr * br - ci * bi;
  float im = cr * bi + ci * br;
  Bwt[(size_t)n * H_ + k] = f2bf(n < P_ ? re : im);
}

__global__ void prep_wc_kernel(const float* __restrict__ Cfre, const float* __restrict__ Cfim,
                               const float* __restrict__ Cbre, const float* __restrict__ Cbim,
                               __bf16* __restrict__ Wct) {
  int o = blockIdx.x * blockDim.x + threadIdx.x;     // over 1024*256
  int n = o >> 8, k = o & 255, q = k & (P_ - 1);
  float v;
  switch (k >> 6) {
    case 0:  v =  2.0f * Cfre[(size_t)n * P_ + q]; break;
    case 1:  v = -2.0f * Cfim[(size_t)n * P_ + q]; break;
    case 2:  v =  2.0f * Cbre[(size_t)n * P_ + q]; break;
    default: v = -2.0f * Cbim[(size_t)n * P_ + q]; break;
  }
  Wct[o] = f2bf(v);
}

// ---------------------------------------------------------------------------
// AdaLN conditioning:  cond = t @ W_proj + b  (tiny GEMM, 16384 dots of 1024)
// cond layout: [4][B][H] = {1+scaleH, shiftH, 1+scaleL, shiftL}
// ---------------------------------------------------------------------------
__global__ void cond_kernel(const float* __restrict__ t,
                            const float* __restrict__ WpH, const float* __restrict__ bpH,
                            const float* __restrict__ WpL, const float* __restrict__ bpL,
                            float* __restrict__ cond) {
  int idx = blockIdx.x * blockDim.x + threadIdx.x;   // 0..16383
  int which = idx >> 13;                             // 0 = H, 1 = L
  int rem = idx & 8191;
  int b = rem >> 11;                                 // 0..3
  int j = rem & 2047;                                // 0..2047
  const float* W  = which ? WpL : WpH;
  const float* bi = which ? bpL : bpH;
  const float* tr = t + (size_t)b * H_;
  float acc = bi[j];
  for (int k = 0; k < H_; ++k) acc += tr[k] * W[(size_t)k * 2048 + j];
  int base = b * H_;
  if (j < H_) cond[(which * 2 + 0) * (B_ * H_) + base + j] = 1.0f + acc;  // scale
  else        cond[(which * 2 + 1) * (B_ * H_) + base + (j - H_)] = acc;  // shift
}

// ---------------------------------------------------------------------------
// AdaLN(zH), AdaLN(zL), pack [zHn | zLn | x] as bf16 rows of hcat [M, 3H].
// ---------------------------------------------------------------------------
__global__ void adaln_pack_kernel(const float* __restrict__ zH, const float* __restrict__ zL,
                                  const float* __restrict__ xin,
                                  const float* __restrict__ cond,
                                  __bf16* __restrict__ hcat) {
  const int m = blockIdx.x;
  const int b = m >> 13;                              // / L_
  const int tid = threadIdx.x;
  const float* zHr = zH + (size_t)m * H_;
  const float* zLr = zL + (size_t)m * H_;
  float sH = 0.f, qH = 0.f, sL = 0.f, qL = 0.f;
  float vH[4], vL[4];
#pragma unroll
  for (int i = 0; i < 4; ++i) {
    float a = zHr[tid + 256 * i]; vH[i] = a; sH += a; qH += a * a;
    float c = zLr[tid + 256 * i]; vL[i] = c; sL += c; qL += c * c;
  }
  for (int off = 16; off > 0; off >>= 1) {
    sH += __shfl_down(sH, off, 32); qH += __shfl_down(qH, off, 32);
    sL += __shfl_down(sL, off, 32); qL += __shfl_down(qL, off, 32);
  }
  __shared__ float red[4][8];
  const int wid = tid >> 5, lane = tid & 31;
  if (lane == 0) { red[0][wid] = sH; red[1][wid] = qH; red[2][wid] = sL; red[3][wid] = qL; }
  __syncthreads();
  if (tid == 0) {
    float a = 0, q = 0, c = 0, d = 0;
    for (int i = 0; i < 8; ++i) { a += red[0][i]; q += red[1][i]; c += red[2][i]; d += red[3][i]; }
    red[0][0] = a; red[1][0] = q; red[2][0] = c; red[3][0] = d;
  }
  __syncthreads();
  const float inv = 1.0f / (float)H_;
  float muH = red[0][0] * inv, varH = red[1][0] * inv - muH * muH;
  float muL = red[2][0] * inv, varL = red[3][0] * inv - muL * muL;
  float rsH = rsqrtf(varH + 1e-5f);
  float rsL = rsqrtf(varL + 1e-5f);
  const float* scH = cond + 0 * (B_ * H_) + b * H_;
  const float* shH = cond + 1 * (B_ * H_) + b * H_;
  const float* scL = cond + 2 * (B_ * H_) + b * H_;
  const float* shL = cond + 3 * (B_ * H_) + b * H_;
  __bf16* row = hcat + (size_t)m * K_INJ;
  const float* xr = xin + (size_t)m * H_;
#pragma unroll
  for (int i = 0; i < 4; ++i) {
    int h = tid + 256 * i;
    row[h]          = f2bf((vH[i] - muH) * rsH * scH[h] + shH[h]);
    row[H_ + h]     = f2bf((vL[i] - muL) * rsL * scL[h] + shL[h]);
    row[2 * H_ + h] = f2bf(xr[h]);
  }
}

// ---------------------------------------------------------------------------
// Bidirectional complex linear-recurrence scan (wave-per-(b,dir,p), chunked)
// ---------------------------------------------------------------------------
__global__ void scan_kernel(const float* __restrict__ Bu, const float* __restrict__ Abar,
                            __bf16* __restrict__ X) {
  const int gid  = blockIdx.x;                  // 0..511
  const int p    = gid & (P_ - 1);
  const int dir  = (gid >> 6) & 1;
  const int b    = gid >> 7;
  const int lane = threadIdx.x & 31;
  const float ar = Abar[2 * p], ai = Abar[2 * p + 1];
  const int CH = L_ / 32;                       // 256
  const size_t base = (size_t)b * L_;

  // pass 1: local end state
  float xr = 0.f, xi = 0.f;
  for (int j = 0; j < CH; ++j) {
    int jj = lane * CH + j;
    int l  = dir ? (L_ - 1 - jj) : jj;
    size_t m = base + l;
    float ur = Bu[m * N_BU + p];
    float ui = Bu[m * N_BU + P_ + p];
    float nr = ar * xr - ai * xi + ur;
    float ni = ar * xi + ai * xr + ui;
    xr = nr; xi = ni;
  }
  // a^256 via 8 complex squarings
  float pr = ar, pi = ai;
  for (int s = 0; s < 8; ++s) { float t = pr * pr - pi * pi; pi = 2.f * pr * pi; pr = t; }
  // inclusive lane scan of (A,E) pairs
  float Ar = pr, Ai = pi, Er = xr, Ei = xi;
  for (int off = 1; off < 32; off <<= 1) {
    float oAr = __shfl_up(Ar, off, 32), oAi = __shfl_up(Ai, off, 32);
    float oEr = __shfl_up(Er, off, 32), oEi = __shfl_up(Ei, off, 32);
    if (lane >= off) {
      float nEr = Ar * oEr - Ai * oEi + Er;
      float nEi = Ar * oEi + Ai * oEr + Ei;
      float nAr = Ar * oAr - Ai * oAi;
      float nAi = Ar * oAi + Ai * oAr;
      Er = nEr; Ei = nEi; Ar = nAr; Ai = nAi;
    }
  }
  // exclusive carry
  float cr = __shfl_up(Er, 1, 32);
  float ci = __shfl_up(Ei, 1, 32);
  if (lane == 0) { cr = 0.f; ci = 0.f; }
  // pass 2: rerun with carry, emit
  xr = cr; xi = ci;
  const int colbase = dir * 2 * P_;
  for (int j = 0; j < CH; ++j) {
    int jj = lane * CH + j;
    int l  = dir ? (L_ - 1 - jj) : jj;
    size_t m = base + l;
    float ur = Bu[m * N_BU + p];
    float ui = Bu[m * N_BU + P_ + p];
    float nr = ar * xr - ai * xi + ur;
    float ni = ar * xi + ai * xr + ui;
    xr = nr; xi = ni;
    X[m * K_OUT + colbase + p]      = f2bf(xr);
    X[m * K_OUT + colbase + P_ + p] = f2bf(xi);
  }
}

// ---------------------------------------------------------------------------
// Host launcher
// ---------------------------------------------------------------------------
extern "C" void kernel_launch(void* const* d_in, const int* in_sizes, int n_in,
                              void* d_out, int out_size, void* d_ws, size_t ws_size,
                              hipStream_t stream) {
  const float* zH      = (const float*)d_in[0];
  const float* zL      = (const float*)d_in[1];
  const float* x       = (const float*)d_in[2];
  const float* t       = (const float*)d_in[3];
  const float* WpH     = (const float*)d_in[4];
  const float* bpH     = (const float*)d_in[5];
  const float* WpL     = (const float*)d_in[6];
  const float* bpL     = (const float*)d_in[7];
  const float* Winj    = (const float*)d_in[8];
  const float* binj    = (const float*)d_in[9];
  const float* Lre     = (const float*)d_in[10];
  const float* Lim     = (const float*)d_in[11];
  const float* Bre     = (const float*)d_in[12];
  const float* Bim     = (const float*)d_in[13];
  const float* Cfre    = (const float*)d_in[14];
  const float* Cfim    = (const float*)d_in[15];
  const float* Cbre    = (const float*)d_in[16];
  const float* Cbim    = (const float*)d_in[17];
  const float* Dvec    = (const float*)d_in[18];
  const float* logdt   = (const float*)d_in[19];

  const size_t MB = 1024ull * 1024ull;
  char* ws = (char*)d_ws;
  float*  cond  = (float*) (ws + 0);                 //  64 KB
  float*  Abar  = (float*) (ws + (64ull << 10));     // 512 B
  __bf16* Wtinj = (__bf16*)(ws + 1 * MB);            //   6 MB  [1024][3072]
  __bf16* Bwt   = (__bf16*)(ws + 8 * MB);            // 256 KB  [128][1024]
  __bf16* Wct   = (__bf16*)(ws + 9 * MB);            // 512 KB  [1024][256]
  __bf16* hcat  = (__bf16*)(ws + 16 * MB);           // 192 MB  [M][3072]
  __bf16* hbuf  = (__bf16*)(ws + 208 * MB);          //  64 MB  [M][1024]
  float*  Bu    = (float*) (ws + 272 * MB);          //  16 MB  [M][128]
  __bf16* X     = (__bf16*)(ws + 288 * MB);          //  16 MB  [M][256]

  // --- weight preparation (tiny) ---
  prep_winj_kernel<<<(H_ * K_INJ) / 256, 256, 0, stream>>>(Winj, Wtinj);
  prep_abar_kernel<<<1, 64, 0, stream>>>(Lre, Lim, logdt, Abar);
  prep_bw_kernel<<<(N_BU * H_) / 256, 256, 0, stream>>>(Bre, Bim, Abar, Lre, Lim, Bwt);
  prep_wc_kernel<<<(H_ * K_OUT) / 256, 256, 0, stream>>>(Cfre, Cfim, Cbre, Cbim, Wct);

  // --- AdaLN conditioning + normalization + pack ---
  cond_kernel<<<(2 * B_ * 2048) / 256, 256, 0, stream>>>(t, WpH, bpH, WpL, bpL, cond);
  adaln_pack_kernel<<<M_, 256, 0, stream>>>(zH, zL, x, cond, hcat);

  // --- inj GEMM: h = hcat @ W_inj + b_inj  (bf16 out), block tile 128x256 ---
  wmma_gemm_kernel<8, true, true, false><<<dim3(M_ / 128, H_ / 256), 256, 0, stream>>>(
      hcat, Wtinj, hbuf, M_, H_, K_INJ, binj, nullptr, nullptr);

  // --- Bu GEMM: Bu = h @ B_bar^T (re|im)  (f32 out), block tile 128x128 ---
  wmma_gemm_kernel<4, false, false, false><<<dim3(M_ / 128, 1), 256, 0, stream>>>(
      hbuf, Bwt, Bu, M_, N_BU, H_, nullptr, nullptr, nullptr);

  // --- bidirectional scans ---
  scan_kernel<<<B_ * 2 * P_, 32, 0, stream>>>(Bu, Abar, X);

  // --- output GEMM: y = X @ Wc^T + D*h  (f32 out to d_out), block tile 128x256 ---
  wmma_gemm_kernel<8, false, false, true><<<dim3(M_ / 128, H_ / 256), 256, 0, stream>>>(
      X, Wct, d_out, M_, H_, K_OUT, nullptr, Dvec, hbuf);
}
